// LinearAttention_21895743275586
// MI455X (gfx1250) — compile-verified, matrix operand order checked by reading, not run on hardware
//
#include <hip/hip_runtime.h>
#include <hip/hip_bf16.h>

#define BATCH   8
#define CH      512
#define HEADS   8
#define DHEAD   64
#define NPIX    4096          // 64*64
#define QKV_CH  1536          // 3*512

typedef __attribute__((ext_vector_type(16))) __bf16         v16bf;
typedef __attribute__((ext_vector_type(8)))  float          v8f;
typedef __attribute__((ext_vector_type(8)))  unsigned int   v8u;
typedef __attribute__((ext_vector_type(4)))  float          v4f;
typedef __attribute__((ext_vector_type(4)))  unsigned short us4;

__device__ __forceinline__ unsigned short f2bf(float f) {
    union { float f; unsigned int u; } c; c.f = f;
    unsigned int u = c.u;
    // round-to-nearest-even fp32 -> bf16
    return (unsigned short)((u + 0x7FFFu + ((u >> 16) & 1u)) >> 16);
}

// ---------------------------------------------------------------------------
// Generic batched bf16 WMMA GEMM:  C[M,N] = alpha*A[M,K] * B[K,N]
//   A row-major fp32 (alpha folded into bf16 conversion of A)
//   B row-major fp32;  if TRANSB, Bmem is [N,K] row-major (logical B = Bmem^T)
//   per-z offsets: off = (z/div)*strideOuter + (z%div)*strideInner
//
// Block: 256 threads (8 waves). Block tile 256(M) x 64(N), K-step 32.
// Each wave computes a 32x64 strip (2 A-frags x 4 B-frags = 8 WMMA / K-step).
//
// LDS holds tiles in *fragment order*: each lane's 16 bf16 fragment is 16
// contiguous ushorts -> compute phase reads 2x ds_load_b128 per fragment.
// Inverse of ISA 7.12.2 16-bit layout:  for K index kk,
//   half = (kk>>3)&1,  j = ((kk&7)>>1) + (kk&16 ? 4 : 0),  slot e = 2j+(kk&1)
// An aligned group of 4 K values maps to 4 contiguous slots in one lane.
// ---------------------------------------------------------------------------
template<bool TRANSB>
__global__ __launch_bounds__(256) void gemm_bf16_wmma(
    const float* __restrict__ A,  long lda, long aDiv, long aOut, long aIn,
    const float* __restrict__ Bm, long ldb, long bDiv, long bOut, long bIn,
    float* __restrict__ C,        long ldc, long cDiv, long cOut, long cIn,
    int M, int N, int Kdim, float alpha)
{
    // 16 A sub-tiles (16 rows each) + 4 B sub-tiles, 32 lanes x 16 ushorts
    __shared__ unsigned short Af[16 * 32 * 16];   // 16 KB
    __shared__ unsigned short Bf[ 4 * 32 * 16];   //  4 KB

    const long z = blockIdx.z;
    A  += (z / aDiv) * aOut + (z % aDiv) * aIn;
    Bm += (z / bDiv) * bOut + (z % bDiv) * bIn;
    C  += (z / cDiv) * cOut + (z % cDiv) * cIn;

    const int tid  = threadIdx.x;
    const int wave = tid >> 5;
    const int lane = tid & 31;
    const int rowBase = blockIdx.x * 256;
    const int colBase = blockIdx.y * 64;
    const int lm   = lane & 15;
    const int half = lane >> 4;

    v8f acc0[4] = {};
    v8f acc1[4] = {};

    for (int k0 = 0; k0 < Kdim; k0 += 32) {
        // ---- stage A tile: 256x32 fp32 -> bf16 fragments (2048 float4) ----
        #pragma unroll
        for (int i = 0; i < 8; ++i) {
            int q  = i * 256 + tid;        // 0..2047
            int r  = q >> 3;               // 0..255
            int kq = (q & 7) * 4;          // 0,4,..,28
            int gr = rowBase + r;
            v4f v = {};
            if (gr < M) v = *(const v4f*)&A[(long)gr * lda + (k0 + kq)];
            int h  = (kq >> 3) & 1;
            int e0 = (((kq & 7) >> 1) + ((kq & 16) ? 4 : 0)) * 2;
            int ld = h * 16 + (r & 15);
            int s  = r >> 4;
            us4 u = { f2bf(v[0] * alpha), f2bf(v[1] * alpha),
                      f2bf(v[2] * alpha), f2bf(v[3] * alpha) };
            *(us4*)&Af[(s * 32 + ld) * 16 + e0] = u;
        }
        // ---- stage B tile: 32x64 fp32 -> bf16 fragments (512 float4) ----
        #pragma unroll
        for (int i = 0; i < 2; ++i) {
            int q = i * 256 + tid;         // 0..511
            if (TRANSB) {
                int nn = q >> 3;           // 0..63 (column of C)
                int kq = (q & 7) * 4;
                v4f v = *(const v4f*)&Bm[(long)(colBase + nn) * ldb + (k0 + kq)];
                int h  = (kq >> 3) & 1;
                int e0 = (((kq & 7) >> 1) + ((kq & 16) ? 4 : 0)) * 2;
                int t  = nn >> 4;
                int ld = h * 16 + (nn & 15);
                us4 u = { f2bf(v[0]), f2bf(v[1]), f2bf(v[2]), f2bf(v[3]) };
                *(us4*)&Bf[(t * 32 + ld) * 16 + e0] = u;
            } else {
                int kk = q >> 4;           // 0..31
                int n4 = (q & 15) * 4;     // 0,4,..,60
                v4f v = *(const v4f*)&Bm[(long)(k0 + kk) * ldb + colBase + n4];
                int h = (kk >> 3) & 1;
                int e = (((kk & 7) >> 1) + ((kk & 16) ? 4 : 0)) * 2 + (kk & 1);
                #pragma unroll
                for (int u = 0; u < 4; ++u) {
                    int nn = n4 + u;
                    int t  = nn >> 4;
                    int ld = h * 16 + (nn & 15);
                    Bf[(t * 32 + ld) * 16 + e] = f2bf(v[u]);
                }
            }
        }
        __syncthreads();

        // ---- fragments: contiguous 32-byte reads (2x ds_load_b128 each) ----
        v16bf a0 = __builtin_bit_cast(v16bf,
                     *(const v8u*)&Af[((wave * 2 + 0) * 32 + lane) * 16]);
        v16bf a1 = __builtin_bit_cast(v16bf,
                     *(const v8u*)&Af[((wave * 2 + 1) * 32 + lane) * 16]);
        v16bf b[4];
        #pragma unroll
        for (int t = 0; t < 4; ++t)
            b[t] = __builtin_bit_cast(v16bf,
                     *(const v8u*)&Bf[(t * 32 + lane) * 16]);

        #pragma unroll
        for (int t = 0; t < 4; ++t) {
            acc0[t] = __builtin_amdgcn_wmma_f32_16x16x32_bf16(
                          false, a0, false, b[t], (short)0, acc0[t], false, false);
            acc1[t] = __builtin_amdgcn_wmma_f32_16x16x32_bf16(
                          false, a1, false, b[t], (short)0, acc1[t], false, false);
        }
        __syncthreads();
    }

    // ---- store: C/D layout — lane n = lane%16, VGPR j -> M = j + 8*half ----
    #pragma unroll
    for (int t = 0; t < 4; ++t) {
        int col = colBase + t * 16 + lm;
        #pragma unroll
        for (int j = 0; j < 8; ++j) {
            int row0 = rowBase + wave * 32 + j + half * 8;
            int row1 = row0 + 16;
            if (row0 < M && col < N) C[(long)row0 * ldc + col] = acc0[t][j];
            if (row1 < M && col < N) C[(long)row1 * ldc + col] = acc1[t][j];
        }
    }
}

// ---------------------------------------------------------------------------
// softmax over d (64 strided values) for q; fold in 1/sqrt(DHEAD)=0.125
// ---------------------------------------------------------------------------
__global__ __launch_bounds__(256) void softmax_q(float* __restrict__ qkv)
{
    int bh = blockIdx.x;
    int b = bh >> 3, h = bh & 7;
    int n = blockIdx.y * 256 + threadIdx.x;
    float* p = qkv + (long)b * QKV_CH * NPIX + (long)h * DHEAD * NPIX + n;

    float vals[64];
    float mx = -1e30f;
    #pragma unroll
    for (int d = 0; d < 64; ++d) { vals[d] = p[(long)d * NPIX]; mx = fmaxf(mx, vals[d]); }
    float s = 0.0f;
    #pragma unroll
    for (int d = 0; d < 64; ++d) { vals[d] = __expf(vals[d] - mx); s += vals[d]; }
    float inv = 0.125f / s;
    #pragma unroll
    for (int d = 0; d < 64; ++d) p[(long)d * NPIX] = vals[d] * inv;
}

// ---------------------------------------------------------------------------
// softmax over N (4096 contiguous floats) for k; one block per (b, channel)
// ---------------------------------------------------------------------------
__global__ __launch_bounds__(256) void softmax_k(float* __restrict__ qkv)
{
    int b = blockIdx.x >> 9;
    int c = blockIdx.x & 511;
    float* p = qkv + (long)b * QKV_CH * NPIX + (long)(CH + c) * NPIX;
    __shared__ float red[256];
    int tid = threadIdx.x;

    float mx = -1e30f;
    for (int i = tid; i < NPIX; i += 256) mx = fmaxf(mx, p[i]);
    red[tid] = mx; __syncthreads();
    for (int s = 128; s > 0; s >>= 1) { if (tid < s) red[tid] = fmaxf(red[tid], red[tid + s]); __syncthreads(); }
    mx = red[0]; __syncthreads();

    float sum = 0.0f;
    for (int i = tid; i < NPIX; i += 256) sum += __expf(p[i] - mx);
    red[tid] = sum; __syncthreads();
    for (int s = 128; s > 0; s >>= 1) { if (tid < s) red[tid] += red[tid + s]; __syncthreads(); }
    float inv = 1.0f / red[0];

    for (int i = tid; i < NPIX; i += 256) p[i] = __expf(p[i] - mx) * inv;
}

// ---------------------------------------------------------------------------
// bias add + LayerNorm over channels (512, stride NPIX), in place on d_out
// ---------------------------------------------------------------------------
__global__ __launch_bounds__(256) void bias_layernorm(float* __restrict__ out,
                                                      const float* __restrict__ bias)
{
    int b = blockIdx.x;
    int n = blockIdx.y * 256 + threadIdx.x;
    float* p = out + (long)b * CH * NPIX + n;

    float s = 0.0f, s2 = 0.0f;
    for (int c = 0; c < CH; ++c) {
        float v = p[(long)c * NPIX] + bias[c];
        s += v; s2 += v * v;
    }
    float mean = s * (1.0f / CH);
    float var  = s2 * (1.0f / CH) - mean * mean;
    float rstd = rsqrtf(var + 1e-5f);
    for (int c = 0; c < CH; ++c) {
        float v = p[(long)c * NPIX] + bias[c];
        p[(long)c * NPIX] = (v - mean) * rstd;
    }
}

// ---------------------------------------------------------------------------
extern "C" void kernel_launch(void* const* d_in, const int* in_sizes, int n_in,
                              void* d_out, int out_size, void* d_ws, size_t ws_size,
                              hipStream_t stream)
{
    const float* x     = (const float*)d_in[0];   // [8,512,64,64]
    const float* w_qkv = (const float*)d_in[1];   // [1536,512]
    const float* w_out = (const float*)d_in[2];   // [512,512]
    const float* b_out = (const float*)d_in[3];   // [512]
    float* out = (float*)d_out;                   // [8,512,64,64]

    float* qkv = (float*)d_ws;                                // 8*1536*4096 f32
    float* ctx = qkv + (long)BATCH * QKV_CH * NPIX;           // 64 * 64*64 f32

    const long nQ = (long)QKV_CH * NPIX;   // per-batch qkv stride
    dim3 blk(256);

    // 1) qkv = w_qkv[1536x512] * x[512x4096]   (batched over B, A shared)
    gemm_bf16_wmma<false><<<dim3(6, 64, 8), blk, 0, stream>>>(
        w_qkv, 512, 1, 0, 0,
        x, NPIX, 1, (long)CH * NPIX, 0,
        qkv, NPIX, 1, nQ, 0,
        QKV_CH, NPIX, CH, 1.0f);

    // 2) softmaxes
    softmax_q<<<dim3(64, 16), blk, 0, stream>>>(qkv);
    softmax_k<<<dim3(BATCH * CH), blk, 0, stream>>>(qkv);

    // 3) ctxT[e,d] = (1/N) * sum_n V[e,n] * K[d,n]    per (b,h): 64 batches
    gemm_bf16_wmma<true><<<dim3(1, 1, 64), blk, 0, stream>>>(
        qkv + 2L * CH * NPIX, NPIX, 8, nQ, (long)DHEAD * NPIX,   // A = V
        qkv + 1L * CH * NPIX, NPIX, 8, nQ, (long)DHEAD * NPIX,   // Bmem = K (transposed use)
        ctx, 64, 1, 64 * 64, 0,
        64, 64, NPIX, 1.0f / (float)NPIX);

    // 4) attn[e,n] = sum_d ctxT[e,d] * Q[d,n]; write into (dead) K region
    gemm_bf16_wmma<false><<<dim3(1, 64, 64), blk, 0, stream>>>(
        ctx, 64, 1, 64 * 64, 0,
        qkv, NPIX, 8, nQ, (long)DHEAD * NPIX,                    // B = Q
        qkv + 1L * CH * NPIX, NPIX, 8, nQ, (long)DHEAD * NPIX,   // C -> K region
        64, NPIX, 64, 1.0f);

    // 5) proj = w_out[512x512] * attn[512x4096]  -> d_out (raw, pre-bias)
    gemm_bf16_wmma<false><<<dim3(2, 64, 8), blk, 0, stream>>>(
        w_out, CH, 1, 0, 0,
        qkv + 1L * CH * NPIX, NPIX, 1, nQ, 0,
        out, NPIX, 1, (long)CH * NPIX, 0,
        CH, NPIX, CH, 1.0f);

    // 6) bias + channel LayerNorm in place
    bias_layernorm<<<dim3(BATCH, 16), blk, 0, stream>>>(out, b_out);
}
